// iRMB_5531917877515
// MI455X (gfx1250) — matrix operand order repaired and checked by reading, not compiled, not run on hardware
//
#include <hip/hip_runtime.h>
#include <cstdint>

// ---------------------------------------------------------------------------
// iRMB block for MI455X (gfx1250, wave32, WMMA bf16, TDM window gather)
// ---------------------------------------------------------------------------

typedef __attribute__((ext_vector_type(8)))  __bf16       v8bf;
typedef __attribute__((ext_vector_type(16))) __bf16       v16bf;
typedef __attribute__((ext_vector_type(8)))  float        v8f;
typedef __attribute__((ext_vector_type(4)))  unsigned int u32x4;
typedef __attribute__((ext_vector_type(4)))  int          i32x4;
typedef __attribute__((ext_vector_type(8)))  int          i32x8;

// ---- LDS layout for the attention kernel (bytes) --------------------------
// region A (73728): Xf32[64][264] -> S[4][64][72] f32 -> ATT[64][264] f32
// region B (36864): Xbf16[64][264] -> P[4][64][72] bf16
#define OFF_A   0
#define OFF_B   73728
#define OFF_Q   110592   // Q bf16 [64][264]
#define OFF_K   144384   // K bf16 [64][264]
#define OFF_V   178176   // Vt f32 [256][72]  (v with scaled residual, c-major)
#define OFF_VB  251904   // Vt bf16 [256][72]
#define LDS_ATTN 288768  // <= 320KB WGP LDS

#define LDS_FFN  135680

// ---------------------------------------------------------------------------
// WMMA fragment helpers (CDNA5 §7.12.2 layouts, wave32)
// A 16x32 bf16: lane(0-15)=row M, k-halves interleaved in 8-element chunks.
// B 32x16 bf16: stored "B^T row-major" (for each N: contiguous K); lane(0-15)=N,
//               lanes 16-31 carry K=8..15,24..31 (as A).  D: lane=N, VGPR r -> M=r(+8 hi).
// ---------------------------------------------------------------------------
static __device__ __forceinline__ v16bf cat8(v8bf lo, v8bf hi) {
  v16bf r;
#pragma unroll
  for (int i = 0; i < 8; ++i) { r[i] = lo[i]; r[i + 8] = hi[i]; }
  return r;
}
static __device__ __forceinline__ v16bf frag_a(const __bf16* base, int stride,
                                               int m0, int k0, int lane) {
  const __bf16* p = base + (m0 + (lane & 15)) * stride + k0 + ((lane & 16) ? 8 : 0);
  return cat8(*(const v8bf*)p, *(const v8bf*)(p + 16));
}
static __device__ __forceinline__ v16bf frag_b(const __bf16* base, int stride,
                                               int n0, int k0, int lane) {
  const __bf16* p = base + (n0 + (lane & 15)) * stride + k0 + ((lane & 16) ? 8 : 0);
  return cat8(*(const v8bf*)p, *(const v8bf*)(p + 16));
}
static __device__ __forceinline__ v8f wmma_bf16(v16bf a, v16bf b, v8f c) {
  return __builtin_amdgcn_wmma_f32_16x16x32_bf16(false, a, false, b, (short)0, c,
                                                 false, false);
}

// ---------------------------------------------------------------------------
// Kernel 0: convert GEMM weights fp32 -> bf16 (one pass, L2-resident after)
// ---------------------------------------------------------------------------
__global__ void irmb_prep(const float* __restrict__ qw, int nq,
                          const float* __restrict__ w1, int n1,
                          const float* __restrict__ w2, int n2,
                          __bf16* __restrict__ qwb, __bf16* __restrict__ w1b,
                          __bf16* __restrict__ w2b) {
  int i = blockIdx.x * 256 + threadIdx.x;
  if (i < nq) qwb[i] = (__bf16)qw[i];
  if (i < n1) w1b[i] = (__bf16)w1[i];
  if (i < n2) w2b[i] = (__bf16)w2[i];
}

// ---------------------------------------------------------------------------
// Kernel 1: per-window fused QKV + attention + depthwise convs + residual.
// One workgroup (256 threads = 8 wave32) per window; 2048 windows.
// ---------------------------------------------------------------------------
__launch_bounds__(256, 1)
__global__ void irmb_attn(const float* __restrict__ x,
                          const __bf16* __restrict__ qkvw,   // [768][256] bf16
                          const float* __restrict__ qkvb,    // [768]
                          const float* __restrict__ w3g,     // [256][9]
                          const float* __restrict__ w5g,     // [256][25]
                          const float* __restrict__ w7g,     // [256][49]
                          float* __restrict__ xo) {
  extern __shared__ char smem[];
  float*  Xf  = (float*)(smem + OFF_A);    // [64][264]
  float*  S   = (float*)(smem + OFF_A);    // [4][64][72]
  float*  ATT = (float*)(smem + OFF_A);    // [64][264]
  __bf16* Xb  = (__bf16*)(smem + OFF_B);   // [64][264]
  __bf16* P   = (__bf16*)(smem + OFF_B);   // [4][64][72]
  __bf16* Qb  = (__bf16*)(smem + OFF_Q);   // [64][264]
  __bf16* Kb  = (__bf16*)(smem + OFF_K);   // [64][264]
  float*  Vt  = (float*)(smem + OFF_V);    // [256][72]
  __bf16* Vtb = (__bf16*)(smem + OFF_VB);  // [256][72]

  const int tid  = threadIdx.x;
  const int lane = tid & 31;
  const int wave = tid >> 5;
  const int n15  = lane & 15;
  const int mhi  = (lane & 16) ? 8 : 0;

  const int wid  = blockIdx.x;   // 2048 = 32 * 8 * 8
  const int bimg = wid >> 6;
  const int n1   = (wid >> 3) & 7;
  const int n2   = wid & 7;

  // ---- Phase 1: TDM gather of the dilated 7x7 window into LDS -------------
  // window pixel (h1,w1) lives at x[bimg, h1*8+n1, w1*8+n2, :]:
  //   3D tile: dim0=c (256, stride 1), dim1=w1 (7, stride 2048),
  //            dim2=h1 (7, stride 114688); TDM LDS padding 8dw per 256dw
  //   makes the LDS row stride 264 floats (bank-conflict friendly).
  if (wave == 0) {
    unsigned long long ga =
        (unsigned long long)(uintptr_t)x +
        4ull * (((unsigned long long)bimg * 56 * 56 +
                 (unsigned long long)n1 * 56 + (unsigned long long)n2) * 256ull);
    u32x4 g0; i32x8 g1; i32x4 g2, g3;
    g0[0] = 1u;                                               // count=1 (valid)
    g0[1] = __builtin_amdgcn_groupstaticsize() + OFF_A;       // lds byte addr
    g0[2] = (unsigned)ga;
    g0[3] = ((unsigned)(ga >> 32) & 0x01FFFFFFu) | (2u << 30);  // type=2
    g1[0] = (2 << 16) | (1 << 20) | (7 << 22) | (7 << 25);  // 4B, pad 8dw/256dw
    g1[1] = (int)(256u << 16);                 // tensor_dim0 = 256
    g1[2] = (int)(7u << 16);                   // tensor_dim1 = 7
    g1[3] = (int)(256u << 16);                 // tile_dim0 = 256
    g1[4] = (int)(7u | (7u << 16));            // tile_dim1=7, tile_dim2=7
    g1[5] = 2048;                              // tensor_dim0_stride
    g1[6] = (int)((114688u & 0xFFFFu) << 16);  // tensor_dim1_stride lo
    g1[7] = (int)(114688u >> 16);              // tensor_dim1_stride hi
    g2[0] = 7; g2[1] = 0; g2[2] = 0; g2[3] = 0;  // tensor_dim2 = 7
    g3[0] = 0; g3[1] = 0; g3[2] = 0; g3[3] = 0;
#if __clang_major__ >= 23
    i32x8 gz = {0, 0, 0, 0, 0, 0, 0, 0};
    __builtin_amdgcn_tensor_load_to_lds(g0, g1, g2, g3, gz, 0);
#else
    __builtin_amdgcn_tensor_load_to_lds(g0, g1, g2, g3, 0);
#endif
    __builtin_amdgcn_s_wait_tensorcnt(0);
  }
  __syncthreads();

  // bf16 copy of X, zero the M-padding rows 49..63
  for (int i = tid; i < 64 * 256; i += 256) {
    int t = i >> 8, c = i & 255;
    float v = (t < 49) ? Xf[t * 264 + c] : 0.f;
    if (t >= 49) Xf[t * 264 + c] = 0.f;
    Xb[t * 264 + c] = (__bf16)v;
  }
  __syncthreads();

  // ---- Phase 2: QKV = X @ Wqkv^T + b ; v = 0.1*v + x (scaled residual) ----
  // 4 M-tiles x 48 N-tiles, K=256 (8 ksteps of 32).  Wave w owns nt%8==w.
  for (int nt = wave; nt < 48; nt += 8) {
    const int obase = nt * 16;
    const int oc = obase + n15;
    const float bv = qkvb[oc];
    for (int mt = 0; mt < 4; ++mt) {
      v8f acc = {0, 0, 0, 0, 0, 0, 0, 0};
#pragma unroll
      for (int ks = 0; ks < 8; ++ks) {
        v16bf a = frag_a(Xb, 264, mt * 16, ks * 32, lane);
        v16bf b = frag_b(qkvw, 256, obase, ks * 32, lane);
        acc = wmma_bf16(a, b, acc);
      }
      if (nt < 16) {                       // Q
#pragma unroll
        for (int r = 0; r < 8; ++r)
          Qb[(mt * 16 + mhi + r) * 264 + oc] = (__bf16)(acc[r] + bv);
      } else if (nt < 32) {                // K
        const int cc = oc - 256;
#pragma unroll
        for (int r = 0; r < 8; ++r)
          Kb[(mt * 16 + mhi + r) * 264 + cc] = (__bf16)(acc[r] + bv);
      } else {                             // V + scaled residual, stored c-major
        const int cc = oc - 512;
#pragma unroll
        for (int r = 0; r < 8; ++r) {
          int t = mt * 16 + mhi + r;
          float v = 0.1f * (acc[r] + bv) + Xf[t * 264 + cc];
          Vt[cc * 72 + t]  = v;
          Vtb[cc * 72 + t] = (__bf16)v;
        }
      }
    }
  }
  __syncthreads();

  // ---- Phase 3: S = Q @ K^T (per head; 2 waves per head) ------------------
  const int head = wave >> 1;
  const int par  = wave & 1;
  for (int mt = par; mt < 4; mt += 2) {
    for (int nt = 0; nt < 4; ++nt) {
      v8f acc = {0, 0, 0, 0, 0, 0, 0, 0};
#pragma unroll
      for (int ks = 0; ks < 2; ++ks) {
        v16bf a = frag_a(Qb, 264, mt * 16, head * 64 + ks * 32, lane);
        v16bf b = frag_b(Kb, 264, nt * 16, head * 64 + ks * 32, lane);
        acc = wmma_bf16(a, b, acc);
      }
      float* Sh = S + head * 64 * 72;
#pragma unroll
      for (int r = 0; r < 8; ++r)
        Sh[(mt * 16 + mhi + r) * 72 + nt * 16 + n15] = acc[r];
    }
  }
  __syncthreads();

  // ---- Phase 4: softmax rows (scale 1/8), P in bf16, cols>=49 zeroed ------
  {
    const int hh = tid >> 6, row = tid & 63;
    const float* Sh = S + hh * 64 * 72 + row * 72;
    __bf16* Ph = P + hh * 64 * 72 + row * 72;
    float sv[49], mx = -1e30f;
#pragma unroll
    for (int j = 0; j < 49; ++j) { float s = Sh[j] * 0.125f; sv[j] = s; mx = fmaxf(mx, s); }
    float sum = 0.f;
#pragma unroll
    for (int j = 0; j < 49; ++j) { float e = __expf(sv[j] - mx); sv[j] = e; sum += e; }
    float inv = 1.f / sum;
#pragma unroll
    for (int j = 0; j < 49; ++j) Ph[j] = (__bf16)(sv[j] * inv);
#pragma unroll
    for (int j = 49; j < 64; ++j) Ph[j] = (__bf16)0.f;
  }
  __syncthreads();

  // ---- Phase 5: attn_out = P @ V (B-frags from c-major Vt, contiguous K) --
  for (int mt = par; mt < 4; mt += 2) {
    for (int nt = 0; nt < 4; ++nt) {
      v8f acc = {0, 0, 0, 0, 0, 0, 0, 0};
#pragma unroll
      for (int ks = 0; ks < 2; ++ks) {
        v16bf a = frag_a(P + head * 64 * 72, 72, mt * 16, ks * 32, lane);
        v16bf b = frag_b(Vtb, 72, head * 64 + nt * 16, ks * 32, lane);
        acc = wmma_bf16(a, b, acc);
      }
#pragma unroll
      for (int r = 0; r < 8; ++r)
        ATT[(mt * 16 + mhi + r) * 264 + head * 64 + nt * 16 + n15] = acc[r];
    }
  }
  __syncthreads();

  // ---- Phase 6: depthwise 3/5/7 convs (f32 VALU) + branch mix + residual --
  {
    const int c = tid;                       // one channel per thread
    const float* vr = Vt + c * 72;           // xc[c, h, w] == vr[h*7+w]
#pragma unroll 1
    for (int t = 0; t < 49; ++t) {
      const int h = t / 7, w = t - 7 * h;
      float s = ATT[t * 264 + c];
      for (int dy = -1; dy <= 1; ++dy) {
        int hh = h + dy; if ((unsigned)hh >= 7u) continue;
        for (int dx = -1; dx <= 1; ++dx) {
          int ww = w + dx; if ((unsigned)ww >= 7u) continue;
          s += vr[hh * 7 + ww] * w3g[c * 9 + (dy + 1) * 3 + (dx + 1)];
        }
      }
      for (int dy = -2; dy <= 2; ++dy) {
        int hh = h + dy; if ((unsigned)hh >= 7u) continue;
        for (int dx = -2; dx <= 2; ++dx) {
          int ww = w + dx; if ((unsigned)ww >= 7u) continue;
          s += vr[hh * 7 + ww] * w5g[c * 25 + (dy + 2) * 5 + (dx + 2)];
        }
      }
      for (int dy = -3; dy <= 3; ++dy) {
        int hh = h + dy; if ((unsigned)hh >= 7u) continue;
        for (int dx = -3; dx <= 3; ++dx) {
          int ww = w + dx; if ((unsigned)ww >= 7u) continue;
          s += vr[hh * 7 + ww] * w7g[c * 49 + (dy + 3) * 7 + (dx + 3)];
        }
      }
      // 0.2 * 0.25 * (attn + c3 + c5 + c7) + shortcut2(v)
      ATT[t * 264 + c] = 0.05f * s + vr[t];
    }
  }
  __syncthreads();

  // ---- coalesced scatter back to the un-windowed layout -------------------
  for (int i = tid; i < 49 * 256; i += 256) {
    int t = i >> 8, c = i & 255;
    int h = t / 7, w = t - 7 * h;
    xo[(((size_t)bimg * 56 + (h * 8 + n1)) * 56 + (w * 8 + n2)) * 256 + c] =
        ATT[t * 264 + c];
  }
}

// ---------------------------------------------------------------------------
// Kernel 2: fused FFN (GEMM1+SiLU+GEMM2) + scaled residual + LayerNorm.
// 64 rows per workgroup (256 threads = 8 waves); 1568 blocks.
// ---------------------------------------------------------------------------
__launch_bounds__(256, 1)
__global__ void irmb_ffn(const float* __restrict__ xo,
                         const __bf16* __restrict__ w1,  // [256][256] bf16
                         const __bf16* __restrict__ w2,  // [256][256] bf16
                         const float* __restrict__ lng,
                         const float* __restrict__ lnb,
                         float* __restrict__ out) {
  extern __shared__ char smem[];
  float*  Af = (float*)(smem);              // [64][264]
  __bf16* Ab = (__bf16*)(smem + 67584);     // [64][264]
  __bf16* Hb = (__bf16*)(smem + 101376);    // [64][264]
  float*  mu = (float*)(smem + 135168);     // [64]
  float*  rs = (float*)(smem + 135424);     // [64]

  const int tid = threadIdx.x, lane = tid & 31, wave = tid >> 5;
  const int n15 = lane & 15, mhi = (lane & 16) ? 8 : 0;
  const size_t r0 = (size_t)blockIdx.x * 64;

  for (int i = tid; i < 64 * 256; i += 256) {
    int t = i >> 8, c = i & 255;
    float v = xo[(r0 + t) * 256 + c];
    Af[t * 264 + c] = v;
    Ab[t * 264 + c] = (__bf16)v;
  }
  __syncthreads();

  // GEMM1 + SiLU  (weights stream from L2-resident bf16 copies)
  for (int nt = wave; nt < 16; nt += 8) {
    for (int mt = 0; mt < 4; ++mt) {
      v8f acc = {0, 0, 0, 0, 0, 0, 0, 0};
#pragma unroll
      for (int ks = 0; ks < 8; ++ks) {
        v16bf a = frag_a(Ab, 264, mt * 16, ks * 32, lane);
        v16bf b = frag_b(w1, 256, nt * 16, ks * 32, lane);
        acc = wmma_bf16(a, b, acc);
      }
#pragma unroll
      for (int r = 0; r < 8; ++r) {
        float y = acc[r];
        Hb[(mt * 16 + mhi + r) * 264 + nt * 16 + n15] =
            (__bf16)(y / (1.f + __expf(-y)));
      }
    }
  }
  __syncthreads();

  // GEMM2 + scaled residual (in place: each element owned by one lane)
  for (int nt = wave; nt < 16; nt += 8) {
    for (int mt = 0; mt < 4; ++mt) {
      v8f acc = {0, 0, 0, 0, 0, 0, 0, 0};
#pragma unroll
      for (int ks = 0; ks < 8; ++ks) {
        v16bf a = frag_a(Hb, 264, mt * 16, ks * 32, lane);
        v16bf b = frag_b(w2, 256, nt * 16, ks * 32, lane);
        acc = wmma_bf16(a, b, acc);
      }
#pragma unroll
      for (int r = 0; r < 8; ++r) {
        int t = mt * 16 + mhi + r, c = nt * 16 + n15;
        Af[t * 264 + c] = 0.5f * acc[r] + Af[t * 264 + c];
      }
    }
  }
  __syncthreads();

  if (tid < 64) {
    float s = 0.f, s2 = 0.f;
    for (int c = 0; c < 256; ++c) {
      float v = Af[tid * 264 + c];
      s += v; s2 += v * v;
    }
    float m = s * (1.f / 256.f);
    float var = s2 * (1.f / 256.f) - m * m;
    mu[tid] = m;
    rs[tid] = rsqrtf(var + 1e-5f);
  }
  __syncthreads();

  for (int i = tid; i < 64 * 256; i += 256) {
    int t = i >> 8, c = i & 255;
    out[(r0 + t) * 256 + c] = (Af[t * 264 + c] - mu[t]) * rs[t] * lng[c] + lnb[c];
  }
}

// ---------------------------------------------------------------------------
// Host launcher.  Workspace layout (needs ~103.5 MB):
//   [0,393216)        qkv_w bf16
//   [393216,524288)   ffn_in_w bf16
//   [524288,655360)   ffn_out_w bf16
//   [655360,...)      xo f32 [32,56,56,256]
// ---------------------------------------------------------------------------
extern "C" void kernel_launch(void* const* d_in, const int* in_sizes, int n_in,
                              void* d_out, int out_size, void* d_ws,
                              size_t ws_size, hipStream_t stream) {
  (void)in_sizes; (void)n_in; (void)out_size; (void)ws_size;
  const float* x     = (const float*)d_in[0];
  const float* qkv_w = (const float*)d_in[1];
  const float* qkv_b = (const float*)d_in[2];
  const float* c3w   = (const float*)d_in[3];
  const float* c5w   = (const float*)d_in[4];
  const float* c7w   = (const float*)d_in[5];
  const float* w1    = (const float*)d_in[6];
  const float* w2    = (const float*)d_in[7];
  const float* lng   = (const float*)d_in[8];
  const float* lnb   = (const float*)d_in[9];

  char* ws = (char*)d_ws;
  __bf16* qkv_wb = (__bf16*)(ws);
  __bf16* w1b    = (__bf16*)(ws + 393216);
  __bf16* w2b    = (__bf16*)(ws + 524288);
  float*  xo     = (float*)(ws + 655360);

  irmb_prep<<<768, 256, 0, stream>>>(qkv_w, 196608, w1, 65536, w2, 65536,
                                     qkv_wb, w1b, w2b);
  irmb_attn<<<2048, 256, LDS_ATTN, stream>>>(x, qkv_wb, qkv_b, c3w, c5w, c7w, xo);
  irmb_ffn<<<1568, 256, LDS_FFN, stream>>>(xo, w1b, w2b, lng, lnb, (float*)d_out);
}
